// QuantWrapper_13692355740454
// MI455X (gfx1250) — compile-verified, matrix operand order checked by reading, not run on hardware
//
#include <hip/hip_runtime.h>

typedef __attribute__((ext_vector_type(16))) _Float16 v16h;
typedef __attribute__((ext_vector_type(8)))  _Float16 v8h;
typedef __attribute__((ext_vector_type(8)))  float    v8f;
typedef __attribute__((ext_vector_type(4)))  int      v4i;

#define M_ROWS 32
#define K_IN   4096
#define N_OUT  16384
#define KSPLIT 8                   // one K-chunk per wave in a block
#define KCHUNK (K_IN / KSPLIT)     // 512
#define KSTEPS (KCHUNK / 32)       // 16 wmma k-steps per wave

// ---- prep: x f32 -> f16 (row-major, tiny: 512 KB read / 256 KB write) ----
__global__ __launch_bounds__(256) void cvt_x_f16(const float* __restrict__ x,
                                                 _Float16* __restrict__ xh) {
  int i = blockIdx.x * 256 + threadIdx.x;
  xh[i] = (_Float16)x[i];
}

// Dequantize 16 int32 codes (two 32B chunks: K = [0..7] and [16..23] relative
// to this lane's half offset) into a 16xf16 A-operand fragment.
__device__ __forceinline__ v16h load_a_dequant(const int* __restrict__ p, int zero) {
  v4i q0 = *(const v4i*)(p);
  v4i q1 = *(const v4i*)(p + 4);
  v4i q2 = *(const v4i*)(p + 16);
  v4i q3 = *(const v4i*)(p + 20);
  v16h a;
  a[0]  = (_Float16)(short)(q0.x - zero);
  a[1]  = (_Float16)(short)(q0.y - zero);
  a[2]  = (_Float16)(short)(q0.z - zero);
  a[3]  = (_Float16)(short)(q0.w - zero);
  a[4]  = (_Float16)(short)(q1.x - zero);
  a[5]  = (_Float16)(short)(q1.y - zero);
  a[6]  = (_Float16)(short)(q1.z - zero);
  a[7]  = (_Float16)(short)(q1.w - zero);
  a[8]  = (_Float16)(short)(q2.x - zero);
  a[9]  = (_Float16)(short)(q2.y - zero);
  a[10] = (_Float16)(short)(q2.z - zero);
  a[11] = (_Float16)(short)(q2.w - zero);
  a[12] = (_Float16)(short)(q3.x - zero);
  a[13] = (_Float16)(short)(q3.y - zero);
  a[14] = (_Float16)(short)(q3.z - zero);
  a[15] = (_Float16)(short)(q3.w - zero);
  return a;
}

// B fragment: 8 f16 at +0 (K=0..7 rel) and 8 f16 at +16 (K=16..23 rel)
__device__ __forceinline__ v16h load_b(const _Float16* __restrict__ p) {
  v8h lo = *(const v8h*)(p);
  v8h hi = *(const v8h*)(p + 16);
  return __builtin_shufflevector(lo, hi, 0, 1, 2, 3, 4, 5, 6, 7,
                                 8, 9, 10, 11, 12, 13, 14, 15);
}

// Block: 256 threads = 8 waves. Block owns 32 output columns (2 o-tiles).
// Wave w owns K-chunk w and computes 4 16x16 partial tiles with 4 WMMAs/k-step.
// LDS reduction over the 8 K-chunks, then scale + bias + store.
__global__ __launch_bounds__(256) void qgemm_wmma(
    const int* __restrict__ wq, const _Float16* __restrict__ xh,
    const float* __restrict__ bias, const int* __restrict__ zerop,
    const float* __restrict__ scalep, float* __restrict__ out) {
  __shared__ float lds[KSPLIT][4][256];   // [kchunk][tile][lane*8+j] = 32 KB

  const int lane = threadIdx.x & 31;
  const int wave = threadIdx.x >> 5;
  const int n16  = lane & 15;
  const int half = lane >> 4;
  const int zero = *zerop;

  const int obase = blockIdx.x * 32;      // 512 blocks cover N_OUT
  const int kb0   = wave * KCHUNK;

  // A: weight rows (lane <-> output column, K contiguous per lane)
  const int* wrow0 = wq + (size_t)(obase + n16) * K_IN + kb0 + half * 8;
  const int* wrow1 = wrow0 + (size_t)16 * K_IN;
  // B: x rows (lane <-> m column of y^T, K contiguous per lane)
  const _Float16* xr0 = xh + (size_t)n16 * K_IN + kb0 + half * 8;
  const _Float16* xr1 = xr0 + (size_t)16 * K_IN;

  v8f c00 = {}, c01 = {}, c10 = {}, c11 = {};

#pragma unroll 2
  for (int ks = 0; ks < KSTEPS; ++ks) {
    const int ko = ks * 32;
    v16h a0 = load_a_dequant(wrow0 + ko, zero);
    v16h a1 = load_a_dequant(wrow1 + ko, zero);
    v16h b0 = load_b(xr0 + ko);
    v16h b1 = load_b(xr1 + ko);
    c00 = __builtin_amdgcn_wmma_f32_16x16x32_f16(false, a0, false, b0,
                                                 (short)0, c00, false, false);
    c01 = __builtin_amdgcn_wmma_f32_16x16x32_f16(false, a0, false, b1,
                                                 (short)0, c01, false, false);
    c10 = __builtin_amdgcn_wmma_f32_16x16x32_f16(false, a1, false, b0,
                                                 (short)0, c10, false, false);
    c11 = __builtin_amdgcn_wmma_f32_16x16x32_f16(false, a1, false, b1,
                                                 (short)0, c11, false, false);
  }

  *(v8f*)&lds[wave][0][lane * 8] = c00;   // (o-tile 0, m-tile 0)
  *(v8f*)&lds[wave][1][lane * 8] = c01;   // (o-tile 0, m-tile 1)
  *(v8f*)&lds[wave][2][lane * 8] = c10;   // (o-tile 1, m-tile 0)
  *(v8f*)&lds[wave][3][lane * 8] = c11;   // (o-tile 1, m-tile 1)
  __syncthreads();

  if (wave < 4) {                         // one reducing wave per output tile
    const int ot = wave >> 1;
    const int mt = wave & 1;
    v8f s = {};
#pragma unroll
    for (int w = 0; w < KSPLIT; ++w)
      s += *(const v8f*)&lds[w][wave][lane * 8];

    const float scale = *scalep;
    // D layout: lane j-th accumulator = row (o) = half*8 + j, col (m) = n16
    const int o = obase + ot * 16 + half * 8;  // 8 consecutive outputs per lane
    const int m = mt * 16 + n16;
    float* yp = out + (size_t)m * N_OUT + o;
    const float* bp = bias + o;
#pragma unroll
    for (int j = 0; j < 8; ++j)
      yp[j] = scale * s[j] + bp[j];
  }
}

extern "C" void kernel_launch(void* const* d_in, const int* in_sizes, int n_in,
                              void* d_out, int out_size, void* d_ws, size_t ws_size,
                              hipStream_t stream) {
  const float* x      = (const float*)d_in[0];
  const int*   wq     = (const int*)d_in[1];
  const int*   zerop  = (const int*)d_in[2];
  const float* scalep = (const float*)d_in[3];
  const float* bias   = (const float*)d_in[4];
  float*       out    = (float*)d_out;
  _Float16*    xh     = (_Float16*)d_ws;   // 32*4096 f16 = 256 KB scratch

  cvt_x_f16<<<(M_ROWS * K_IN) / 256, 256, 0, stream>>>(x, xh);
  qgemm_wmma<<<N_OUT / 32, 256, 0, stream>>>(wq, xh, bias, zerop, scalep, out);
}
